// DifferentialMarlinAttention_50723563766134
// MI455X (gfx1250) — matrix operand order repaired and checked by reading, not compiled, not run on hardware
//
#include <hip/hip_runtime.h>

typedef __attribute__((ext_vector_type(16))) __bf16 v16bf;
typedef __attribute__((ext_vector_type(8)))  __bf16 v8bf;
typedef __attribute__((ext_vector_type(8)))  float  v8f;

#define N_HEADS 16
#define SEQLEN  2048
#define HDIM    128
#define KTILE   32

// D = A*B + C, bf16 inputs, f32 accum (v_wmma_f32_16x16x32_bf16)
static __device__ __forceinline__ v8f wmma_bf16(v16bf a, v16bf b, v8f c) {
  return __builtin_amdgcn_wmma_f32_16x16x32_bf16(false, a, false, b, (short)0, c,
                                                 false, false);
}

// A-fragment (16x32 bf16): lane holds row (lane&15); elem e -> k = e + 16*(lane>>4)
// => two contiguous 8-elem runs at base+16*hl and base+16*hl+8 (16B-aligned LDS b128 loads)
static __device__ __forceinline__ v16bf load_afrag(const __bf16* __restrict__ base, int hl) {
  const v8bf u = *(const v8bf*)(base + 16 * hl);
  const v8bf w = *(const v8bf*)(base + 16 * hl + 8);
  v16bf r;
#pragma unroll
  for (int j = 0; j < 8; ++j) { r[j] = u[j]; r[j + 8] = w[j]; }
  return r;
}

// Online-softmax update for one 32-key chunk of S^T (C layout: lane = query col,
// elem j = key row j+8*hl; sa = keys 0..15, sb = keys 16..31 of the chunk).
// Rescales acc, updates (m,l), returns P^T as a B-layout fragment
// (B layout: lane = query col, elem e -> key n = e + 16*hl).
static __device__ __forceinline__ v16bf softmax_update(v8f sa, v8f sb, float& m, float& l,
                                                       v8f* acc, int hl) {
  float cm = -3.0e38f;
#pragma unroll
  for (int j = 0; j < 8; ++j) cm = fmaxf(cm, fmaxf(sa[j], sb[j]));
  cm = fmaxf(cm, __shfl_xor(cm, 16, 32));
  const float nm   = fmaxf(m, cm);
  const float corr = __expf(m - nm);  // first iter: exp(-inf)=0
  m = nm;
  float e0[8], e1[8];
  float rs = 0.f;
#pragma unroll
  for (int j = 0; j < 8; ++j) {
    e0[j] = __expf(sa[j] - nm);
    e1[j] = __expf(sb[j] - nm);
    rs += e0[j] + e1[j];
  }
  rs += __shfl_xor(rs, 16, 32);
  l = l * corr + rs;
#pragma unroll
  for (int i = 0; i < 8; ++i)
#pragma unroll
    for (int j = 0; j < 8; ++j) acc[i][j] *= corr;
  v16bf p;
#pragma unroll
  for (int j = 0; j < 8; ++j) {
    const float t0 = __shfl_xor(e0[j], 16, 32);  // partner lane's sa-exp
    const float t1 = __shfl_xor(e1[j], 16, 32);  // partner lane's sb-exp
    p[j]     = (__bf16)(hl ? t1 : e0[j]);  // n = j      (hl=0) | n = 16+j (hl=1)
    p[j + 8] = (__bf16)(hl ? e1[j] : t0);  // n = 8+j    (hl=0) | n = 24+j (hl=1)
  }
  return p;
}

static __device__ __forceinline__ void stage_row16(const float* __restrict__ g,
                                                   __bf16* __restrict__ dst) {
#pragma unroll
  for (int half = 0; half < 2; ++half) {
    const float4 a = *(const float4*)(g + half * 8);
    const float4 b = *(const float4*)(g + half * 8 + 4);
    v8bf s;
    s[0] = (__bf16)a.x; s[1] = (__bf16)a.y; s[2] = (__bf16)a.z; s[3] = (__bf16)a.w;
    s[4] = (__bf16)b.x; s[5] = (__bf16)b.y; s[6] = (__bf16)b.z; s[7] = (__bf16)b.w;
    *(v8bf*)(dst + half * 8) = s;  // ds_store_b128
  }
}

__global__ __launch_bounds__(256)
void DifferentialMarlinAttention_50723563766134_kernel(
    const float* __restrict__ q1g, const float* __restrict__ k1g,
    const float* __restrict__ vg,  const float* __restrict__ q2g,
    const float* __restrict__ k2g, const float* __restrict__ lamlog,
    float* __restrict__ outg) {
  __shared__ __bf16 ldsK1[KTILE * HDIM];  // [key][d] bf16
  __shared__ __bf16 ldsK2[KTILE * HDIM];  // [key][d] bf16
  __shared__ __bf16 ldsVT[HDIM * KTILE];  // [d][key] bf16 (V transposed)

  const int h    = blockIdx.x;       // 0..15 query head
  const int qblk = blockIdx.y;       // 0..15 tile of 128 queries
  const int kvh  = h >> 2;           // GQA 4:1
  const int tid  = threadIdx.x;
  const int wave = tid >> 5;
  const int lane = tid & 31;
  const int lo   = lane & 15;        // query column owned by this lane
  const int hl   = lane >> 4;

  const int   qrow  = qblk * 128 + wave * 16 + lo;
  const float scale = 0.08838834764831845f;  // 1/sqrt(128)

  // Q fragments in B-layout: lane holds Q row `qrow`; elem e -> d = c*32 + e + 16*hl
  v16bf q1f[4], q2f[4];
  {
    const float* p1 = q1g + ((size_t)h * SEQLEN + qrow) * HDIM;
    const float* p2 = q2g + ((size_t)h * SEQLEN + qrow) * HDIM;
#pragma unroll
    for (int c = 0; c < 4; ++c) {
      const int d0 = c * 32 + 16 * hl;
#pragma unroll
      for (int e = 0; e < 16; ++e) {
        q1f[c][e] = (__bf16)(p1[d0 + e] * scale);
        q2f[c][e] = (__bf16)(p2[d0 + e] * scale);
      }
    }
  }

  v8f acc1[8], acc2[8];  // O^T: acc[dc] lane=query col, elem j -> d = dc*16 + j + 8*hl
#pragma unroll
  for (int i = 0; i < 8; ++i)
#pragma unroll
    for (int j = 0; j < 8; ++j) { acc1[i][j] = 0.f; acc2[i][j] = 0.f; }
  float m1 = -3.0e38f, m2 = -3.0e38f, l1 = 0.f, l2 = 0.f;

  const int srow = tid >> 3;        // 0..31 : key row staged by this thread
  const int scol = (tid & 7) * 16;  // 16-float segment

  for (int kb = 0; kb < SEQLEN; kb += KTILE) {
    // ---- cooperative stage: K1,K2 row-major bf16; V transposed bf16 ----
    {
      const size_t gb = ((size_t)kvh * SEQLEN + kb + srow) * HDIM + scol;
      stage_row16(k1g + gb, ldsK1 + srow * HDIM + scol);
      stage_row16(k2g + gb, ldsK2 + srow * HDIM + scol);
#pragma unroll
      for (int i = 0; i < 16; i += 4) {
        const float4 a = *(const float4*)(vg + gb + i);
        ldsVT[(scol + i + 0) * KTILE + srow] = (__bf16)a.x;
        ldsVT[(scol + i + 1) * KTILE + srow] = (__bf16)a.y;
        ldsVT[(scol + i + 2) * KTILE + srow] = (__bf16)a.z;
        ldsVT[(scol + i + 3) * KTILE + srow] = (__bf16)a.w;
      }
    }
    __syncthreads();

    // ---- S^T = K * Q^T over d (4 chained k=32 WMMAs), two 16-key subtiles ----
    v8f s1a, s1b, s2a, s2b;
#pragma unroll
    for (int j = 0; j < 8; ++j) { s1a[j] = s1b[j] = s2a[j] = s2b[j] = 0.f; }
#pragma unroll
    for (int c = 0; c < 4; ++c) {
      const v16bf ka1 = load_afrag(ldsK1 + (lo)      * HDIM + c * 32, hl);
      const v16bf kb1 = load_afrag(ldsK1 + (16 + lo) * HDIM + c * 32, hl);
      const v16bf ka2 = load_afrag(ldsK2 + (lo)      * HDIM + c * 32, hl);
      const v16bf kb2 = load_afrag(ldsK2 + (16 + lo) * HDIM + c * 32, hl);
      s1a = wmma_bf16(ka1, q1f[c], s1a);
      s1b = wmma_bf16(kb1, q1f[c], s1b);
      s2a = wmma_bf16(ka2, q2f[c], s2a);
      s2b = wmma_bf16(kb2, q2f[c], s2b);
    }

    // ---- online softmax + P^T fragments ----
    const v16bf p1 = softmax_update(s1a, s1b, m1, l1, acc1, hl);
    const v16bf p2 = softmax_update(s2a, s2b, m2, l2, acc2, hl);

    // ---- O^T += V^T * P^T (8 d-chunks of 16, shared V fragment) ----
#pragma unroll
    for (int dc = 0; dc < 8; ++dc) {
      const v16bf vf = load_afrag(ldsVT + (dc * 16 + lo) * KTILE, hl);
      acc1[dc] = wmma_bf16(vf, p1, acc1[dc]);
      acc2[dc] = wmma_bf16(vf, p2, acc2[dc]);
    }
    __syncthreads();
  }

  // ---- epilogue: out = attn1/l1 - exp(lambda)*attn2/l2 ----
  const float lam = __expf(lamlog[0]);
  const float r1  = 1.0f / l1;
  const float r2  = lam / l2;
  float* op = outg + ((size_t)h * SEQLEN + qrow) * HDIM;
#pragma unroll
  for (int dc = 0; dc < 8; ++dc)
#pragma unroll
    for (int j = 0; j < 8; ++j)
      op[dc * 16 + 8 * hl + j] = acc1[dc][j] * r1 - acc2[dc][j] * r2;
}

extern "C" void kernel_launch(void* const* d_in, const int* in_sizes, int n_in,
                              void* d_out, int out_size, void* d_ws, size_t ws_size,
                              hipStream_t stream) {
  const float* q1 = (const float*)d_in[0];
  const float* k1 = (const float*)d_in[1];
  const float* v  = (const float*)d_in[2];
  const float* q2 = (const float*)d_in[3];
  const float* k2 = (const float*)d_in[4];
  const float* ll = (const float*)d_in[5];
  float* out = (float*)d_out;

  dim3 grid(N_HEADS, SEQLEN / 128);  // (head, 128-query tile)
  DifferentialMarlinAttention_50723563766134_kernel<<<grid, 256, 0, stream>>>(
      q1, k1, v, q2, k2, ll, out);
}
